// Disentangler_14224931684908
// MI455X (gfx1250) — matrix operand order repaired and checked by reading, not compiled, not run on hardware
//
#include <hip/hip_runtime.h>
#include <hip/hip_bf16.h>

// ---- problem constants (from reference: T,TOK,P,N,D,L,C = 16,16384,8192,32768,128,8,16)
constexpr int T   = 16;
constexpr int TOK = 16384;
constexpr int P   = 8192;
constexpr int N   = 32768;
constexpr int D   = 128;
constexpr int L   = 8;
constexpr int C   = 16;
constexpr int LC  = L * C;            // 128
constexpr int GROUP_SHIFT = 12;       // N/L = 4096 = 2^12
constexpr int BLOCKS_PER_T = 32;      // accum blocks per timestamp
constexpr int ROWS_PER_BLOCK = P / BLOCKS_PER_T;  // 256
constexpr int ROWS_PER_WAVE  = ROWS_PER_BLOCK / 8; // 32

typedef __attribute__((ext_vector_type(16))) _Float16 v16h;
typedef __attribute__((ext_vector_type(8)))  float    v8f;

// ---------------------------------------------------------------------------
// Kernel A: gather rows of x at node_pos, LayerNorm(D=128), pool D->C (mean of
// 8), bin by group l = node_ids>>12, write per-block partial sums to workspace.
// One wave per row: lane i loads row[4i..4i+3] -> one coalesced 512B burst.
// Deterministic: register bins -> LDS -> fixed-order block reduction.
// ---------------------------------------------------------------------------
__global__ __launch_bounds__(256) void accum_kernel(
    const float* __restrict__ x,
    const float* __restrict__ g_enc, const float* __restrict__ b_enc,
    const int*   __restrict__ node_pos, const int* __restrict__ node_ids,
    float* __restrict__ partials /* [T*BLOCKS_PER_T][LC] */) {
  __shared__ float part[8][C][L];   // [wave][c][l]

  const int tid  = threadIdx.x;
  const int wave = tid >> 5;
  const int lane = tid & 31;
  const int t    = blockIdx.x >> 5;        // / BLOCKS_PER_T
  const int blk  = blockIdx.x & 31;        // % BLOCKS_PER_T

  // per-lane LN affine params for d = 4*lane .. 4*lane+3 (hoisted out of loop)
  const float4 ge = ((const float4*)g_enc)[lane];
  const float4 be = ((const float4*)b_enc)[lane];

  float bins[L];
#pragma unroll
  for (int ll = 0; ll < L; ++ll) bins[ll] = 0.0f;

  const int pBase = blk * ROWS_PER_BLOCK + wave * ROWS_PER_WAVE;
  for (int i = 0; i < ROWS_PER_WAVE; ++i) {
    const int p   = pBase + i;
    const int pos = node_pos[t * P + p];    // uniform across wave (broadcast)
    const int nid = node_ids[t * P + p];
    const int l   = nid >> GROUP_SHIFT;

    const float4 v = *(const float4*)(x + ((size_t)t * TOK + (size_t)pos) * D + lane * 4);

    float s  = v.x + v.y + v.z + v.w;
    float sq = v.x * v.x + v.y * v.y + v.z * v.z + v.w * v.w;
#pragma unroll
    for (int m = 16; m >= 1; m >>= 1) {     // wave32 butterfly allreduce
      s  += __shfl_xor(s,  m);
      sq += __shfl_xor(sq, m);
    }
    const float mean = s * (1.0f / 128.0f);
    const float var  = sq * (1.0f / 128.0f) - mean * mean;
    const float rinv = rsqrtf(var + 1e-5f);

    const float n0 = (v.x - mean) * rinv * ge.x + be.x;
    const float n1 = (v.y - mean) * rinv * ge.y + be.y;
    const float n2 = (v.z - mean) * rinv * ge.z + be.z;
    const float n3 = (v.w - mean) * rinv * ge.w + be.w;

    // pooled channel c = lane>>1 gets the sum of 8 consecutive d's
    const float partial = n0 + n1 + n2 + n3;
    const float pairsum = partial + __shfl_xor(partial, 1);

    // even lanes own channel c = lane>>1; accumulate into register bin[l]
#pragma unroll
    for (int ll = 0; ll < L; ++ll)
      bins[ll] += ((ll == l) & ((lane & 1) == 0)) ? pairsum : 0.0f;
  }

  if ((lane & 1) == 0) {
    const int c = lane >> 1;
#pragma unroll
    for (int ll = 0; ll < L; ++ll) part[wave][c][ll] = bins[ll];
  }
  __syncthreads();

  if (tid < LC) {           // j = l*16 + c  == tid
    const int l = tid >> 4, c = tid & 15;
    float s = 0.0f;
#pragma unroll
    for (int w = 0; w < 8; ++w) s += part[w][c][l];   // fixed order
    partials[(size_t)blockIdx.x * LC + tid] = s;
  }
}

// ---------------------------------------------------------------------------
// Kernel B: reduce partials -> comp, LN(g_fin) over 128 per t, LN(g_dec) over
// 16 per (t,l) -> DN[128,16], then OutRows[128,128] = DN x A2^T via WMMA.
// A2^T[c,d]=1 iff c==d>>3 (exact in f16); DN split hi/lo -> 2 chained f16
// WMMAs with f32 accumulation for fp32-accurate results.
// ---------------------------------------------------------------------------
__global__ __launch_bounds__(256) void finalize_wmma_kernel(
    const float* __restrict__ partials,
    const float* __restrict__ g_fin, const float* __restrict__ b_fin,
    const float* __restrict__ g_dec, const float* __restrict__ b_dec,
    float* __restrict__ outRows /* [T*L][D] */) {
  __shared__ float comp[T * LC];    // 2048 f
  __shared__ float cln [T * LC];    // 2048 f
  __shared__ float dn  [T * L * C]; // 2048 f

  const int tid = threadIdx.x;
  constexpr float scale = 1.0f / (8.0f * 4096.0f);  // pool mean * group mean

  // deterministic reduction over the 32 accumulation blocks per t
  for (int i = tid; i < T * LC; i += 256) {
    const int t = i >> 7, j = i & 127;
    float s = 0.0f;
#pragma unroll
    for (int b = 0; b < BLOCKS_PER_T; ++b)
      s += partials[((size_t)t * BLOCKS_PER_T + b) * LC + j];
    comp[i] = s * scale;
  }
  __syncthreads();

  // encode_final_layer_norm over 128 per t
  if (tid < T) {
    const int t = tid;
    float s = 0.0f, sq = 0.0f;
    for (int j = 0; j < LC; ++j) { const float v = comp[t * LC + j]; s += v; sq += v * v; }
    const float m = s * (1.0f / LC);
    const float rinv = rsqrtf(sq * (1.0f / LC) - m * m + 1e-5f);
    for (int j = 0; j < LC; ++j)
      cln[t * LC + j] = (comp[t * LC + j] - m) * rinv * g_fin[j] + b_fin[j];
  }
  __syncthreads();

  // decode_norm over C=16 per (t,l) row
  if (tid < T * L) {
    const int row = tid;  // cln[row*16 + c] is exactly comp_ln[t, l*16+c]
    float s = 0.0f, sq = 0.0f;
    for (int c = 0; c < C; ++c) { const float v = cln[row * C + c]; s += v; sq += v * v; }
    const float m = s * (1.0f / C);
    const float rinv = rsqrtf(sq * (1.0f / C) - m * m + 1e-5f);
    for (int c = 0; c < C; ++c)
      dn[row * C + c] = (cln[row * C + c] - m) * rinv * g_dec[c] + b_dec[c];
  }
  __syncthreads();

  // ---- WMMA: OutRows = DN(128x16) x A2^T(16x128), K padded 16->32 ----
  const int wave  = tid >> 5;
  const int lane  = tid & 31;
  const int mlane = lane & 15;
  const int mrow  = wave * 16 + mlane;

  // A layout (16-bit A 16x32, ISA 7.12.2): lanes 0-15 halfs 0..7 hold K=0..7,
  // lanes 16-31 halfs 0..7 hold K=8..15; halfs 8..15 (K=16..31) zero-padded.
  v16h a_hi, a_lo;
#pragma unroll
  for (int j = 0; j < 16; ++j) { a_hi[j] = (_Float16)0.0f; a_lo[j] = (_Float16)0.0f; }
#pragma unroll
  for (int j = 0; j < 8; ++j) {
    const int c = (lane < 16) ? j : (8 + j);
    const float v = dn[mrow * C + c];
    const _Float16 hi = (_Float16)v;
    a_hi[j] = hi;
    a_lo[j] = (_Float16)(v - (float)hi);
  }

#pragma unroll
  for (int nt = 0; nt < 8; ++nt) {
    // B layout: lanes 0-15 halfs 0..7 = B[K=0..7, n]; lanes 16-31 halfs 0..7 = B[K=8..15, n]; halfs 8..15 = K pad (0).
    // B[c, n] = A2^T[c, nt*16+n] = (c == (nt*16+n)>>3)
    v16h b;
    const int dcol = nt * 16 + mlane;
#pragma unroll
    for (int j = 0; j < 16; ++j)
      b[j] = (_Float16)(((j < 8) && ((dcol >> 3) == ((lane >> 4) * 8 + j))) ? 1.0f : 0.0f);

    v8f acc = {};
    acc = __builtin_amdgcn_wmma_f32_16x16x32_f16(false, a_lo, false, b,
                                                 (short)0, acc, false, false);
    acc = __builtin_amdgcn_wmma_f32_16x16x32_f16(false, a_hi, false, b,
                                                 (short)0, acc, false, false);

    // D layout (32-bit C/D 16x16): lane<16 -> (m=r, n=lane); lane>=16 -> (m=8+r)
#pragma unroll
    for (int r = 0; r < 8; ++r) {
      const int m = (lane < 16) ? r : (8 + r);
      outRows[(wave * 16 + m) * D + nt * 16 + mlane] = acc[r];
    }
  }
}

// ---------------------------------------------------------------------------
// Kernel C: broadcast the 128 distinct rows to the full [T,N,D] output.
// Pure streaming b128 stores; row table (64KB) stays hot in L2.
// ---------------------------------------------------------------------------
__global__ __launch_bounds__(256) void broadcast_kernel(
    const float* __restrict__ outRows, float* __restrict__ out) {
  constexpr int TOTAL4 = T * N * (D / 4);   // 16,777,216 float4 stores
  const float4* rows4 = (const float4*)outRows;
  float4* out4 = (float4*)out;
  const int stride = gridDim.x * blockDim.x;
  for (int q = blockIdx.x * blockDim.x + threadIdx.x; q < TOTAL4; q += stride) {
    const int d4 = q & 31;                  // D/4 = 32 float4 per row
    const int n  = (q >> 5) & (N - 1);
    const int t  = q >> 20;                 // N*(D/4) = 2^20 per t
    out4[q] = rows4[(((t << 3) + (n >> GROUP_SHIFT)) << 5) + d4];
  }
}

// ---------------------------------------------------------------------------
extern "C" void kernel_launch(void* const* d_in, const int* in_sizes, int n_in,
                              void* d_out, int out_size, void* d_ws, size_t ws_size,
                              hipStream_t stream) {
  const float* x        = (const float*)d_in[0];
  const float* g_enc    = (const float*)d_in[1];
  const float* b_enc    = (const float*)d_in[2];
  const float* g_fin    = (const float*)d_in[3];
  const float* b_fin    = (const float*)d_in[4];
  const float* g_dec    = (const float*)d_in[5];
  const float* b_dec    = (const float*)d_in[6];
  const int*   node_pos = (const int*)d_in[7];
  const int*   node_ids = (const int*)d_in[8];
  // d_in[9] = num_nodes (N), baked into constants above

  float* partials = (float*)d_ws;                               // T*32*128 f = 256 KB
  float* outRows  = partials + (size_t)T * BLOCKS_PER_T * LC;   // 128*128 f  =  64 KB
  float* out      = (float*)d_out;

  accum_kernel<<<T * BLOCKS_PER_T, 256, 0, stream>>>(
      x, g_enc, b_enc, node_pos, node_ids, partials);
  finalize_wmma_kernel<<<1, 256, 0, stream>>>(
      partials, g_fin, b_fin, g_dec, b_dec, outRows);
  broadcast_kernel<<<8192, 256, 0, stream>>>(outRows, out);
}
